// AttnBlock_8821862826455
// MI455X (gfx1250) — matrix operand order repaired and checked by reading, not compiled, not run on hardware
//
#include <hip/hip_runtime.h>

// Problem constants (match reference)
#define B_ 2
#define S_ 2048
#define E_ 768
#define H_ 12
#define D_ 64
#define F_ 3072
#define MROWS 4096  // B_*S_
#define BK 64       // GEMM K-step per LDS stage

typedef __attribute__((ext_vector_type(16))) _Float16 v16h;
typedef __attribute__((ext_vector_type(8)))  _Float16 v8h;
typedef __attribute__((ext_vector_type(8)))  float    v8f;

// gfx1250 async global->LDS path (guarded; fallback = load + ds_store)
#if defined(__HIP_DEVICE_COMPILE__) && \
    __has_builtin(__builtin_amdgcn_global_load_async_to_lds_b128)
#define USE_ASYNC_LDS 1
#else
#define USE_ASYNC_LDS 0
#endif

static __device__ __forceinline__ v8f wmma_f16f32(v16h a, v16h b, v8f c) {
  // D = A(16x32 f16) * B(32x16 f16) + C(16x16 f32)
  return __builtin_amdgcn_wmma_f32_16x16x32_f16(false, a, false, b, (short)0, c,
                                                false, false);
}
static __device__ __forceinline__ v16h cat8(v8h lo, v8h hi) {
  return __builtin_shufflevector(lo, hi, 0,1,2,3,4,5,6,7,8,9,10,11,12,13,14,15);
}
static __device__ __forceinline__ float rmax16(float x) {
#pragma unroll
  for (int o = 1; o < 16; o <<= 1) x = fmaxf(x, __shfl_xor(x, o, 16));
  return x;
}
static __device__ __forceinline__ float rsum16(float x) {
#pragma unroll
  for (int o = 1; o < 16; o <<= 1) x += __shfl_xor(x, o, 16);
  return x;
}

#if USE_ASYNC_LDS
// Builtin signature (from hipcc diagnostic): param 0 is
// 'int __vector(4) __device__ *' i.e. v4i in addrspace(1); dst is LDS (AS3).
typedef __attribute__((vector_size(16))) int v4i_;
typedef __attribute__((address_space(1))) v4i_ g_v4i;
typedef __attribute__((address_space(3))) v4i_ l_v4i;

static __device__ __forceinline__ void async_b128(const _Float16* gsrc,
                                                  _Float16* lds_dst) {
  __builtin_amdgcn_global_load_async_to_lds_b128((g_v4i*)gsrc, (l_v4i*)lds_dst,
                                                 0, 0);
}
static __device__ __forceinline__ void wait_async0() {
#if __has_builtin(__builtin_amdgcn_s_wait_asynccnt)
  __builtin_amdgcn_s_wait_asynccnt(0);
#else
  asm volatile("s_wait_asynccnt 0" ::: "memory");
#endif
}
#endif

// ---------------------------------------------------------------------------
// Weight convert + transpose: WT[n*K + k] = (f16) W[(n/DD)*s_h + k*s_k + n%DD]
// Plain [K x N] row-major weight: DD = N, s_h = 0, s_k = N.
// Wq/Wv [H,E,D]: DD = D, s_h = E*D, s_k = D (col index n = h*64+d).
// ---------------------------------------------------------------------------
__global__ __launch_bounds__(256) void convT_kernel(
    const float* __restrict__ W, _Float16* __restrict__ WT,
    int K, int N, int DD, long s_h, long s_k)
{
  long idx = (long)blockIdx.x * 256 + threadIdx.x;
  if (idx >= (long)K * N) return;
  long n = idx / K;
  long k = idx - n * K;
  long src = (n / DD) * s_h + k * s_k + (n % DD);
  WT[idx] = (_Float16)W[src];
}

// ---------------------------------------------------------------------------
// LayerNorm over E=768, one row per 256-thread block, f16 output.
// ---------------------------------------------------------------------------
__global__ __launch_bounds__(256) void ln_kernel(
    const float* __restrict__ x, const float* __restrict__ g,
    const float* __restrict__ bb, _Float16* __restrict__ o)
{
  __shared__ float red[256];
  const int row = blockIdx.x, tid = threadIdx.x;
  const float* xr = x + (long)row * E_;
  float v0 = xr[tid], v1 = xr[tid + 256], v2 = xr[tid + 512];
  red[tid] = v0 + v1 + v2;
  __syncthreads();
  for (int o2 = 128; o2 > 0; o2 >>= 1) {
    if (tid < o2) red[tid] += red[tid + o2];
    __syncthreads();
  }
  const float mu = red[0] * (1.0f / 768.0f);
  __syncthreads();
  float d0 = v0 - mu, d1 = v1 - mu, d2 = v2 - mu;
  red[tid] = d0 * d0 + d1 * d1 + d2 * d2;
  __syncthreads();
  for (int o2 = 128; o2 > 0; o2 >>= 1) {
    if (tid < o2) red[tid] += red[tid + o2];
    __syncthreads();
  }
  const float rs = rsqrtf(red[0] * (1.0f / 768.0f) + 1e-5f);
  _Float16* orow = o + (long)row * E_;
  orow[tid]       = (_Float16)(d0 * rs * g[tid]       + bb[tid]);
  orow[tid + 256] = (_Float16)(d1 * rs * g[tid + 256] + bb[tid + 256]);
  orow[tid + 512] = (_Float16)(d2 * rs * g[tid + 512] + bb[tid + 512]);
}

// ---------------------------------------------------------------------------
// Generic WMMA GEMM: out = act(A[MxK] @ B[KxN] + bias) (+resid)
// A: f16 row-major [M][K]; BT: f16 transposed [N][K].
// Block tile 64x128, 8 waves, wave tile 32x32, K step 64 via LDS (24KB).
// Global->LDS staging uses GLOBAL_LOAD_ASYNC_TO_LDS_B128 when available.
// outH mode 0: row-major [M][N]; mode 1: vT layout [B,H,D,S].
// ---------------------------------------------------------------------------
__global__ __launch_bounds__(256) void gemm_wmma(
    const _Float16* __restrict__ A, const _Float16* __restrict__ BT,
    const float* __restrict__ bias, const float* __restrict__ resid,
    float* __restrict__ outF, _Float16* __restrict__ outH,
    int M, int N, int K, int relu, int mode)
{
  __shared__ alignas(128) _Float16 As[64][BK];
  __shared__ alignas(128) _Float16 Bs[128][BK];
  const int tid = threadIdx.x;
  const int lane = tid & 31, w = tid >> 5;
  const int hi = lane >> 4, nn = lane & 15;
  const int row0 = blockIdx.x * 64, col0 = blockIdx.y * 128;
  const int wm = (w >> 2) * 32, wn = (w & 3) * 32;

  v8f cz = {};
  v8f acc[2][2];
  acc[0][0] = cz; acc[0][1] = cz; acc[1][0] = cz; acc[1][1] = cz;

  const int ar = tid >> 2, ac = (tid & 3) * 16;  // A stage: 64 rows x 64, 16 h/thr
  const int bn = tid >> 1, bc = (tid & 1) * 32;  // B stage: 128 rows x 64, 32 h/thr

  for (int k0 = 0; k0 < K; k0 += BK) {
    const _Float16* ga = A + (long)(row0 + ar) * K + k0 + ac;
    const _Float16* gb = BT + (long)(col0 + bn) * K + k0 + bc;
    if (k0 + BK < K) {  // prefetch next K tile into GL2 (global_prefetch_b8)
      __builtin_prefetch((const void*)(ga + BK), 0, 0);
      __builtin_prefetch((const void*)(gb + BK), 0, 0);
    }
#if USE_ASYNC_LDS
    async_b128(ga,      &As[ar][ac]);
    async_b128(ga + 8,  &As[ar][ac + 8]);
    async_b128(gb,      &Bs[bn][bc]);
    async_b128(gb + 8,  &Bs[bn][bc + 8]);
    async_b128(gb + 16, &Bs[bn][bc + 16]);
    async_b128(gb + 24, &Bs[bn][bc + 24]);
    wait_async0();
#else
    *(v16h*)&As[ar][ac] = *(const v16h*)ga;
    *(v16h*)&Bs[bn][bc] = *(const v16h*)gb;
    *(v16h*)&Bs[bn][bc + 16] = *(const v16h*)(gb + 16);
#endif
    __syncthreads();
#pragma unroll
    for (int kk = 0; kk < 2; ++kk) {
      v16h af[2], bf[2];
#pragma unroll
      for (int i = 0; i < 2; ++i) {
        // A-frag (16-bit 16x32): lane row = nn, 16 contiguous K halves at hi*16.
        const _Float16* ap = &As[wm + 16 * i + nn][kk * 32 + hi * 16];
        af[i] = *(const v16h*)ap;
      }
#pragma unroll
      for (int j = 0; j < 2; ++j) {
        // B-frag (32x16): lane col = nn, 16 contiguous K halves at hi*16.
        bf[j] = *(const v16h*)&Bs[wn + 16 * j + nn][kk * 32 + hi * 16];
      }
#pragma unroll
      for (int i = 0; i < 2; ++i)
#pragma unroll
        for (int j = 0; j < 2; ++j)
          acc[i][j] = wmma_f16f32(af[i], bf[j], acc[i][j]);
    }
    __syncthreads();
  }

#pragma unroll
  for (int i = 0; i < 2; ++i) {
#pragma unroll
    for (int j = 0; j < 2; ++j) {
#pragma unroll
      for (int e = 0; e < 8; ++e) {
        const int r = row0 + wm + 16 * i + e + 8 * hi;
        const int cidx = col0 + wn + 16 * j + nn;
        float val = acc[i][j][e] + bias[cidx];
        if (relu) val = fmaxf(val, 0.0f);
        if (resid) val += resid[(long)r * N + cidx];
        if (outF) outF[(long)r * N + cidx] = val;
        if (outH) {
          long oi = (mode == 0)
                        ? ((long)r * N + cidx)
                        : (((long)(r >> 11) * E_ + cidx) * (long)S_ + (r & 2047));
          outH[oi] = (_Float16)val;
        }
      }
    }
  }
}

// ---------------------------------------------------------------------------
// Flash attention, one wave (32 threads) per 16-row q block per (b,h).
// k = q (reference bug reproduced). scores = q k^T / 8, causal + input mask,
// online softmax; P re-layout via 1KB LDS; out = P @ V with vT [B,H,D,S].
// All-masked rows (l==0) fall back to out = v[row] (diag=1 edge case).
// ---------------------------------------------------------------------------
__global__ __launch_bounds__(32) void attn_wmma(
    const _Float16* __restrict__ q, const _Float16* __restrict__ vT,
    const int* __restrict__ mask, _Float16* __restrict__ outH)
{
  __shared__ alignas(64) _Float16 pl[16][32];
  const int lane = threadIdx.x;
  const int nn = lane & 15, hi = lane >> 4;
  const int blk = blockIdx.x;
  const int rb = blk & 127;     // 128 row-blocks of 16 (S=2048)
  const int bh = blk >> 7;
  const int h = bh % H_;
  const int b = bh / H_;
  const int s0 = rb * 16;
  const float scale = 0.125f;
  const float NEGINF = -__builtin_inff();

  // q fragments for d 0..31 and 32..63 (A layout; rows contiguous in d).
  const _Float16* qrow = q + (long)(b * S_ + s0 + nn) * E_ + h * D_;
  v16h aq0 = *(const v16h*)(qrow + hi * 16);
  v16h aq1 = *(const v16h*)(qrow + 32 + hi * 16);

  float mr[8], lr[8];
  v8f cz = {};
  v8f oacc[4];
#pragma unroll
  for (int j = 0; j < 4; ++j) oacc[j] = cz;
#pragma unroll
  for (int i = 0; i < 8; ++i) { mr[i] = NEGINF; lr[i] = 0.0f; }

  const int nChunk = (s0 + 47) / 32;  // cover t in [0, s0+15]
  for (int c = 0; c < nChunk; ++c) {
    const int t0 = c * 32;
    v8f sc0, sc1;
    {
      const _Float16* kp = q + (long)(b * S_ + t0 + nn) * E_ + h * D_;
      v16h bk0 = *(const v16h*)(kp + hi * 16);
      v16h bk1 = *(const v16h*)(kp + 32 + hi * 16);
      sc0 = wmma_f16f32(aq0, bk0, cz);
      sc0 = wmma_f16f32(aq1, bk1, sc0);
    }
    {
      const _Float16* kp = q + (long)(b * S_ + t0 + 16 + nn) * E_ + h * D_;
      v16h bk0 = *(const v16h*)(kp + hi * 16);
      v16h bk1 = *(const v16h*)(kp + 32 + hi * 16);
      sc1 = wmma_f16f32(aq0, bk0, cz);
      sc1 = wmma_f16f32(aq1, bk1, sc1);
    }
    const int imA = mask[b * S_ + t0 + nn];
    const int imB = mask[b * S_ + t0 + 16 + nn];
#pragma unroll
    for (int i = 0; i < 8; ++i) {
      const int row = s0 + i + 8 * hi;
      float sA = sc0[i] * scale;
      float sB = sc1[i] * scale;
      sA = ((t0 + nn) > row || imA == 0) ? NEGINF : sA;
      sB = ((t0 + 16 + nn) > row || imB == 0) ? NEGINF : sB;
      const float mnew = fmaxf(mr[i], rmax16(fmaxf(sA, sB)));
      const float alpha = (mr[i] > NEGINF) ? __expf(mr[i] - mnew) : 0.0f;
      const float pA = (sA > NEGINF) ? __expf(sA - mnew) : 0.0f;
      const float pB = (sB > NEGINF) ? __expf(sB - mnew) : 0.0f;
      lr[i] = lr[i] * alpha + rsum16(pA + pB);
      mr[i] = mnew;
#pragma unroll
      for (int j = 0; j < 4; ++j) oacc[j][i] *= alpha;
      pl[i + 8 * hi][nn]      = (_Float16)pA;
      pl[i + 8 * hi][16 + nn] = (_Float16)pB;
    }
    // LDS from same wave is in-order; explicit split-counter wait for safety.
    asm volatile("s_wait_dscnt 0" ::: "memory");
    v16h ap = *(const v16h*)&pl[nn][hi * 16];
#pragma unroll
    for (int j = 0; j < 4; ++j) {
      const _Float16* vp =
          vT + ((long)((b * H_ + h) * D_ + 16 * j + nn)) * (long)S_ + t0 + hi * 16;
      v16h bv = *(const v16h*)vp;
      oacc[j] = wmma_f16f32(ap, bv, oacc[j]);
    }
  }

#pragma unroll
  for (int i = 0; i < 8; ++i) {
    const int row = s0 + i + 8 * hi;
    const float inv = (lr[i] > 0.0f) ? 1.0f / lr[i] : 0.0f;
#pragma unroll
    for (int j = 0; j < 4; ++j) {
      const int dl = 16 * j + nn;
      float val = oacc[j][i] * inv;
      const float vfall =
          (float)vT[((long)((b * H_ + h) * D_ + dl)) * (long)S_ + row];
      val = (lr[i] > 0.0f) ? val : vfall;  // all-masked-row edge case
      outH[(long)(b * S_ + row) * E_ + h * D_ + dl] = (_Float16)val;
    }
  }
}

// ---------------------------------------------------------------------------
extern "C" void kernel_launch(void* const* d_in, const int* in_sizes, int n_in,
                              void* d_out, int out_size, void* d_ws, size_t ws_size,
                              hipStream_t stream) {
  const float* x   = (const float*)d_in[0];
  const float* Wq  = (const float*)d_in[1];
  const float* bq  = (const float*)d_in[2];
  const float* Wv  = (const float*)d_in[3];
  const float* bv  = (const float*)d_in[4];
  const float* Wo  = (const float*)d_in[5];
  const float* bo  = (const float*)d_in[6];
  const float* g1  = (const float*)d_in[7];
  const float* be1 = (const float*)d_in[8];
  const float* W1  = (const float*)d_in[9];
  const float* b1  = (const float*)d_in[10];
  const float* W2  = (const float*)d_in[11];
  const float* b2  = (const float*)d_in[12];
  const float* W3  = (const float*)d_in[13];
  const float* b3  = (const float*)d_in[14];
  const float* g2  = (const float*)d_in[15];
  const float* be2 = (const float*)d_in[16];
  const int*   im  = (const int*)d_in[17];

  char* cur = (char*)d_ws;
  auto alloc = [&](size_t bytes) -> void* {
    void* p = (void*)cur;
    cur += (bytes + 255) & ~(size_t)255;
    return p;
  };
  _Float16* xln = (_Float16*)alloc((size_t)MROWS * E_ * 2);
  _Float16* WqT = (_Float16*)alloc((size_t)E_ * E_ * 2);
  _Float16* WvT = (_Float16*)alloc((size_t)E_ * E_ * 2);
  _Float16* WoT = (_Float16*)alloc((size_t)E_ * E_ * 2);
  _Float16* W1T = (_Float16*)alloc((size_t)E_ * F_ * 2);
  _Float16* W2T = (_Float16*)alloc((size_t)F_ * F_ * 2);
  _Float16* W3T = (_Float16*)alloc((size_t)F_ * E_ * 2);
  _Float16* qh  = (_Float16*)alloc((size_t)MROWS * E_ * 2);
  _Float16* vTb = (_Float16*)alloc((size_t)MROWS * E_ * 2);
  _Float16* ah  = (_Float16*)alloc((size_t)MROWS * E_ * 2);
  float*    y   = (float*)alloc((size_t)MROWS * E_ * 4);
  _Float16* yln = (_Float16*)alloc((size_t)MROWS * E_ * 2);
  _Float16* h1  = (_Float16*)alloc((size_t)MROWS * F_ * 2);
  _Float16* h2  = (_Float16*)alloc((size_t)MROWS * F_ * 2);

  auto cgrid = [](long n) { return (int)((n + 255) / 256); };
  // Weight conversion (f32 -> f16 transposed [N][K])
  convT_kernel<<<cgrid((long)E_ * E_), 256, 0, stream>>>(Wq, WqT, E_, E_, D_, (long)E_ * D_, (long)D_);
  convT_kernel<<<cgrid((long)E_ * E_), 256, 0, stream>>>(Wv, WvT, E_, E_, D_, (long)E_ * D_, (long)D_);
  convT_kernel<<<cgrid((long)E_ * E_), 256, 0, stream>>>(Wo, WoT, E_, E_, E_, 0L, (long)E_);
  convT_kernel<<<cgrid((long)E_ * F_), 256, 0, stream>>>(W1, W1T, E_, F_, F_, 0L, (long)F_);
  convT_kernel<<<cgrid((long)F_ * F_), 256, 0, stream>>>(W2, W2T, F_, F_, F_, 0L, (long)F_);
  convT_kernel<<<cgrid((long)F_ * E_), 256, 0, stream>>>(W3, W3T, F_, E_, E_, 0L, (long)E_);

  // LN1
  ln_kernel<<<MROWS, 256, 0, stream>>>(x, g1, be1, xln);

  dim3 gE(MROWS / 64, E_ / 128);
  dim3 gF(MROWS / 64, F_ / 128);
  // q = xln @ Wq + bq (f16 out, row-major); k = q (reference bug)
  gemm_wmma<<<gE, 256, 0, stream>>>(xln, WqT, bq, nullptr, nullptr, qh,
                                    MROWS, E_, E_, 0, 0);
  // v (f16 out in vT [B,H,D,S] layout for attention B-fragments)
  gemm_wmma<<<gE, 256, 0, stream>>>(xln, WvT, bv, nullptr, nullptr, vTb,
                                    MROWS, E_, E_, 0, 1);
  // attention: B*H*(S/16) waves
  attn_wmma<<<B_ * H_ * (S_ / 16), 32, 0, stream>>>(qh, vTb, im, ah);
  // y = x + attn @ Wo + bo (f32)
  gemm_wmma<<<gE, 256, 0, stream>>>(ah, WoT, bo, x, y, nullptr,
                                    MROWS, E_, E_, 0, 0);
  // LN2
  ln_kernel<<<MROWS, 256, 0, stream>>>(y, g2, be2, yln);
  // FFN
  gemm_wmma<<<gF, 256, 0, stream>>>(yln, W1T, b1, nullptr, nullptr, h1,
                                    MROWS, F_, E_, 1, 0);
  gemm_wmma<<<gF, 256, 0, stream>>>(h1, W2T, b2, nullptr, nullptr, h2,
                                    MROWS, F_, F_, 1, 0);
  gemm_wmma<<<gE, 256, 0, stream>>>(h2, W3T, b3, y, (float*)d_out, nullptr,
                                    MROWS, E_, F_, 0, 0);
}
